// xLSTM_85298050499334
// MI455X (gfx1250) — matrix operand order repaired and checked by reading, not compiled, not run on hardware
//
#include <hip/hip_runtime.h>
#include <hip/hip_bf16.h>

typedef __attribute__((ext_vector_type(16))) __bf16 v16bf;
typedef __attribute__((ext_vector_type(8)))  __bf16 v8bf;
typedef __attribute__((ext_vector_type(8)))  float  v8f;
typedef __attribute__((ext_vector_type(4)))  unsigned int v4u;
typedef __attribute__((ext_vector_type(8)))  int v8i;
typedef __attribute__((ext_vector_type(4)))  int v4i;

#define GATE_NT 4   // 16-wide n-tiles per wave in the generic GEMM

// ---------------------------------------------------------------------------
// CDNA5 async copy: global -> LDS, 16 bytes per lane, tracked by ASYNCcnt
// ---------------------------------------------------------------------------
__device__ inline void async_copy_b128(unsigned int lds_off, const void* gptr) {
    asm volatile("global_load_async_to_lds_b128 %0, %1, off"
                 :: "v"(lds_off), "v"((unsigned long long)(uintptr_t)gptr)
                 : "memory");
}
__device__ inline void wait_asynccnt0() {
    asm volatile("s_wait_asynccnt 0" ::: "memory");
}

// ---------------------------------------------------------------------------
// CDNA5 Tensor Data Mover: one-instruction 2D tile DMA global -> LDS.
// D# packing per ISA 8.3/8.4: group0 = {count, lds_addr, global_addr, type},
// group1 = {data_size, tensor dims, tile dims, dim0 stride}.  2D tile of
// 16-bit elements: tile_w elems per row, tile_h rows, row stride in elems.
// This toolchain exposes the 6-arg builtin: (g0, g1, g2, g3, g1ext, cpol).
// ---------------------------------------------------------------------------
__device__ inline void tdm_load_2d_bf16(unsigned int lds_off, const void* gbase,
                                        unsigned long long row_stride_elems,
                                        unsigned int tile_w, unsigned int tile_h) {
    unsigned long long ga = (unsigned long long)(uintptr_t)gbase;
    v4u g0;
    g0.x = 1u;                                        // count=1 (valid user D#)
    g0.y = lds_off;                                   // lds_addr  [63:32]
    g0.z = (unsigned int)(ga & 0xFFFFFFFFu);          // global_addr[31:0]  -> bits 95:64
    g0.w = (unsigned int)(ga >> 32) | (2u << 30);     // global_addr[56:32] | type=2
    v8i g1;
    g1[0] = (int)(1u << 16);                          // wg_mask=0, data_size=1 (2 bytes)
    g1[1] = (int)((tile_w & 0xFFFFu) << 16);          // tensor_dim0[15:0] @ bits 63:48
    g1[2] = (int)(((tile_w >> 16) & 0xFFFFu) |        // tensor_dim0[31:16]
                  ((tile_h & 0xFFFFu) << 16));        // tensor_dim1[15:0]
    g1[3] = (int)(((tile_h >> 16) & 0xFFFFu) |        // tensor_dim1[31:16]
                  ((tile_w & 0xFFFFu) << 16));        // tile_dim0
    g1[4] = (int)(tile_h & 0xFFFFu);                  // tile_dim1 (tile_dim2 = 0)
    g1[5] = (int)(unsigned int)(row_stride_elems & 0xFFFFFFFFull);  // dim0_stride[31:0]
    g1[6] = (int)((row_stride_elems >> 32) & 0xFFFFull);            // dim0_stride[47:32]
    g1[7] = 0;                                        // tensor_dim1_stride = 0 (2D)
    v4i z4 = {0, 0, 0, 0};
    v8i z8 = {0, 0, 0, 0, 0, 0, 0, 0};
    __builtin_amdgcn_tensor_load_to_lds(g0, g1, z4, z4, z8, 0);
}
__device__ inline void wait_tensorcnt0() {
    __builtin_amdgcn_s_wait_tensorcnt(0);
}

// ---------------------------------------------------------------------------
// helpers
// ---------------------------------------------------------------------------
__device__ inline v16bf pack_a_bf16(const float* __restrict__ arow, int half) {
    const float4* q0 = (const float4*)(arow + half * 8);
    const float4* q1 = (const float4*)(arow + 16 + half * 8);
    float4 f0 = q0[0], f1 = q0[1], f2 = q1[0], f3 = q1[1];
    v16bf a;
    a[0]  = (__bf16)f0.x; a[1]  = (__bf16)f0.y; a[2]  = (__bf16)f0.z; a[3]  = (__bf16)f0.w;
    a[4]  = (__bf16)f1.x; a[5]  = (__bf16)f1.y; a[6]  = (__bf16)f1.z; a[7]  = (__bf16)f1.w;
    a[8]  = (__bf16)f2.x; a[9]  = (__bf16)f2.y; a[10] = (__bf16)f2.z; a[11] = (__bf16)f2.w;
    a[12] = (__bf16)f3.x; a[13] = (__bf16)f3.y; a[14] = (__bf16)f3.z; a[15] = (__bf16)f3.w;
    return a;
}
// A fragment from LDS bf16 row: two 16B ds_load_b128 chunks
__device__ inline v16bf lds_a_frag(const __bf16* arow, int kc, int half) {
    v8bf lo = *(const v8bf*)(arow + kc + half * 8);
    v8bf hi = *(const v8bf*)(arow + kc + 16 + half * 8);
    return __builtin_shufflevector(lo, hi, 0,1,2,3,4,5,6,7,8,9,10,11,12,13,14,15);
}

__device__ inline float sigm(float x) { return 1.0f / (1.0f + __expf(-x)); }
__device__ inline float clampf(float x, float lo, float hi) { return fminf(fmaxf(x, lo), hi); }

// ---------------------------------------------------------------------------
// generic WMMA GEMM (used for the big parallel input projection):
// Out[r][n] = sum_k A[r][k] * W[n][k] + bias[n]
// ---------------------------------------------------------------------------
__global__ __launch_bounds__(128) void wmma_gemm_kernel(
    const float* __restrict__ A0, long long lda0,
    int K,
    const __bf16* __restrict__ W, long long ldw,
    const float* __restrict__ bias,
    float* __restrict__ Out, long long ldo,
    int mTiles, int nGroups)
{
    int wave   = blockIdx.x * (blockDim.x >> 5) + (threadIdx.x >> 5);
    int lane   = threadIdx.x & 31;
    int mt     = wave % mTiles;
    int ng     = wave / mTiles;
    if (ng >= nGroups) return;
    int lane16 = lane & 15;
    int half   = lane >> 4;
    int m      = mt * 16 + lane16;

    v8f acc[GATE_NT] = {};

    for (int kc = 0; kc < K; kc += 32) {
        const float* arow = A0 + (size_t)m * lda0 + kc;
        if (kc + 32 < K)
            __builtin_prefetch((const void*)(arow + 32), 0, 3);
        v16bf a = pack_a_bf16(arow, half);
#pragma unroll
        for (int j = 0; j < GATE_NT; ++j) {
            int n = (ng * GATE_NT + j) * 16 + lane16;
            const __bf16* wr = W + (size_t)n * ldw + kc;
            v8bf blo = *(const v8bf*)(wr + half * 8);
            v8bf bhi = *(const v8bf*)(wr + 16 + half * 8);
            v16bf b = __builtin_shufflevector(blo, bhi, 0,1,2,3,4,5,6,7,8,9,10,11,12,13,14,15);
            acc[j] = __builtin_amdgcn_wmma_f32_16x16x32_bf16(
                false, a, false, b, (short)0, acc[j], false, false);
        }
    }
#pragma unroll
    for (int j = 0; j < GATE_NT; ++j) {
        int n   = (ng * GATE_NT + j) * 16 + lane16;
        float bv = bias ? bias[n] : 0.0f;
#pragma unroll
        for (int r = 0; r < 8; ++r) {
            int row = mt * 16 + half * 8 + r;
            Out[(size_t)row * ldo + n] = acc[j][r] + bv;
        }
    }
}

// ---------------------------------------------------------------------------
// per-step gate GEMM:  gates = [x_t | h] @ Wcat^T + bias
// Block = 256 thr (8 waves) = 4 m-tiles x 8 n-tiles.  The x-slice (64x512
// bf16, row stride T*H) and h (64x512 bf16) are DMA'd into LDS by the
// Tensor Data Mover (two tensor_load_to_lds ops issued by wave 0), then all
// waves read A fragments via ds_load_b128.
// ---------------------------------------------------------------------------
__global__ __launch_bounds__(256) void gate_gemm_kernel(
    const __bf16* __restrict__ xbf,    // (B, T, H) bf16 layer input
    const __bf16* __restrict__ hbf,    // (64, 512) bf16 hidden state
    const __bf16* __restrict__ W,      // (6656, 1024) bf16
    const float* __restrict__ bias,    // (6656)
    float* __restrict__ gates,         // (64, 6656)
    int t, int T)
{
    const int G = 6656, H = 512, K = 1024;
    extern __shared__ __bf16 sA[];     // sX: 64x512, then sH: 64x512 (128 KB)
    __bf16* sX = sA;
    __bf16* sH = sA + 64 * 512;
    int tid = threadIdx.x;

    // ---- TDM stage: comb = [x_t | h] as two 2D tiles ----
    if (tid < 32) {
        unsigned int offX = (unsigned int)(uintptr_t)sX;
        unsigned int offH = (unsigned int)(uintptr_t)sH;
        tdm_load_2d_bf16(offX, xbf + (size_t)t * H, (unsigned long long)T * H, 512u, 64u);
        tdm_load_2d_bf16(offH, hbf, 512ull, 512u, 64u);
        wait_tensorcnt0();
    }
    __syncthreads();

    int wv = tid >> 5, lane = tid & 31, lane16 = lane & 15, half = lane >> 4;
    int mt = wv & 3, nh = wv >> 2;
    const __bf16* arowX = sX + (size_t)(mt * 16 + lane16) * 512;
    const __bf16* arowH = sH + (size_t)(mt * 16 + lane16) * 512;

    v8f acc[4] = {};
    for (int kc = 0; kc < K; kc += 32) {
        const __bf16* ar = (kc < 512) ? arowX : arowH;
        v16bf a = lds_a_frag(ar, kc & 511, half);
#pragma unroll
        for (int j = 0; j < 4; ++j) {
            int n = (blockIdx.x * 8 + nh * 4 + j) * 16 + lane16;
            const __bf16* wr = W + (size_t)n * K + kc;
            v8bf blo = *(const v8bf*)(wr + half * 8);
            v8bf bhi = *(const v8bf*)(wr + 16 + half * 8);
            v16bf b = __builtin_shufflevector(blo, bhi, 0,1,2,3,4,5,6,7,8,9,10,11,12,13,14,15);
            acc[j] = __builtin_amdgcn_wmma_f32_16x16x32_bf16(
                false, a, false, b, (short)0, acc[j], false, false);
        }
    }
#pragma unroll
    for (int j = 0; j < 4; ++j) {
        int n = (blockIdx.x * 8 + nh * 4 + j) * 16 + lane16;
        float bv = bias[n];
#pragma unroll
        for (int r = 0; r < 8; ++r) {
            int row = mt * 16 + half * 8 + r;
            gates[(size_t)row * G + n] = acc[j][r] + bv;
        }
    }
}

// ---------------------------------------------------------------------------
// cell update + LayerNorm over M=2048: one block per batch row.
// Writes c (f32, for next step's f*c) and cbf (bf16, for cell2's GEMM).
// ---------------------------------------------------------------------------
__global__ __launch_bounds__(256) void cellln_kernel(
    const float* __restrict__ gates, float* __restrict__ cstate,
    __bf16* __restrict__ cbf,
    const float* __restrict__ cw, const float* __restrict__ cb)
{
    const int G = 6656, M = 2048;
    int b = blockIdx.x, tid = threadIdx.x;
    __shared__ float sred[256];
    const float* g = gates + (size_t)b * G;
    float* c = cstate + (size_t)b * M;

    float u[8];
    float psum = 0.0f;
#pragma unroll
    for (int e = 0; e < 8; ++e) {
        int j = tid + 256 * e;
        float iv = __expf(clampf(g[j], -10.0f, 10.0f));
        float fv = sigm(g[2048 + j]);
        float zv = tanhf(g[4608 + j]);
        float un = clampf(fv * c[j] + iv * zv, -100.0f, 100.0f);
        u[e] = un; psum += un;
    }
    sred[tid] = psum; __syncthreads();
    for (int s = 128; s > 0; s >>= 1) { if (tid < s) sred[tid] += sred[tid + s]; __syncthreads(); }
    float mu = sred[0] / (float)M;
    __syncthreads();
    float p2 = 0.0f;
#pragma unroll
    for (int e = 0; e < 8; ++e) { float d = u[e] - mu; p2 += d * d; }
    sred[tid] = p2; __syncthreads();
    for (int s = 128; s > 0; s >>= 1) { if (tid < s) sred[tid] += sred[tid + s]; __syncthreads(); }
    float rstd = rsqrtf(sred[0] / (float)M + 1e-5f);
#pragma unroll
    for (int e = 0; e < 8; ++e) {
        int j = tid + 256 * e;
        float cn = (u[e] - mu) * rstd * cw[j] + cb[j];
        c[j] = cn;
        cbf[(size_t)b * M + j] = (__bf16)cn;
    }
}

// ---------------------------------------------------------------------------
// m = c_new @ Wm^T + bm (K=2048), h = LN(sigmoid(o_pre)*tanh(m)).
// Block = 16 batch rows; c-tile (16x2048 bf16 = 64KB) async-staged in LDS
// via GLOBAL_LOAD_ASYNC_TO_LDS_B128 (ASYNCcnt path).
// Writes h as bf16 (next step's gate GEMM) and ys[t] (f32).
// ---------------------------------------------------------------------------
__global__ __launch_bounds__(256) void cell2_kernel(
    const __bf16* __restrict__ cbf,    // (64, 2048) bf16
    const float* __restrict__ gates,   // (64, 6656), o-gate at +4096
    const __bf16* __restrict__ Wm,     // (512, 2048) bf16
    const float* __restrict__ bm,
    const float* __restrict__ hw, const float* __restrict__ hb,
    __bf16* __restrict__ hbf,          // (64, 512) bf16
    float* __restrict__ yout,          // (B, T, H)
    int t, int T)
{
    const int G = 6656, M = 2048, H = 512;
    extern __shared__ __bf16 sC[];     // 16 x 2048 bf16 = 64 KB (dynamic)
    __shared__ float sm[16][516];
    __shared__ float red[16][17];
    __shared__ float mu_s[16], rs_s[16];

    int b0  = blockIdx.x * 16;
    int tid = threadIdx.x;

    // ---- async stage c-tile ----
    unsigned int sbase = (unsigned int)(uintptr_t)(&sC[0]);
#pragma unroll
    for (int it = 0; it < 16; ++it) {
        int chunk = tid + it * 256;          // 4096 x 16B
        int row   = chunk >> 8;              // 256 chunks per 2048-col row
        int cidx  = chunk & 255;
        const __bf16* src = cbf + (size_t)(b0 + row) * M + cidx * 8;
        async_copy_b128(sbase + (unsigned int)chunk * 16, src);
    }
    wait_asynccnt0();
    __syncthreads();

    int wv = tid >> 5, lane = tid & 31, lane16 = lane & 15, half = lane >> 4;
    const __bf16* arow = sC + (size_t)lane16 * M;

    v8f acc[4] = {};
    for (int kc = 0; kc < M; kc += 32) {
        v16bf a = lds_a_frag(arow, kc, half);
#pragma unroll
        for (int j = 0; j < 4; ++j) {
            int n = (wv * 4 + j) * 16 + lane16;
            const __bf16* wr = Wm + (size_t)n * M + kc;
            v8bf blo = *(const v8bf*)(wr + half * 8);
            v8bf bhi = *(const v8bf*)(wr + 16 + half * 8);
            v16bf bv = __builtin_shufflevector(blo, bhi, 0,1,2,3,4,5,6,7,8,9,10,11,12,13,14,15);
            acc[j] = __builtin_amdgcn_wmma_f32_16x16x32_bf16(
                false, a, false, bv, (short)0, acc[j], false, false);
        }
    }
#pragma unroll
    for (int j = 0; j < 4; ++j) {
        int n = (wv * 4 + j) * 16 + lane16;
        float bias = bm[n];
#pragma unroll
        for (int r = 0; r < 8; ++r)
            sm[half * 8 + r][n] = acc[j][r] + bias;
    }
    __syncthreads();

    // h = LN(sigmoid(o_pre) * tanh(m)) per row (16 rows x 16 threads/row)
    int row = tid >> 4;
    int c0  = tid & 15;
    const float* grow = gates + (size_t)(b0 + row) * G + 4096;
    float vals[32];
    float psum = 0.0f;
#pragma unroll
    for (int i = 0; i < 32; ++i) {
        int cc = c0 + 16 * i;
        float v = sigm(grow[cc]) * tanhf(sm[row][cc]);
        vals[i] = v; psum += v;
    }
    red[row][c0] = psum; __syncthreads();
    if (c0 == 0) { float s = 0; for (int i = 0; i < 16; ++i) s += red[row][i]; mu_s[row] = s / (float)H; }
    __syncthreads();
    float mu = mu_s[row];
    float p2 = 0.0f;
#pragma unroll
    for (int i = 0; i < 32; ++i) { float d = vals[i] - mu; p2 += d * d; }
    red[row][c0] = p2; __syncthreads();
    if (c0 == 0) { float s = 0; for (int i = 0; i < 16; ++i) s += red[row][i]; rs_s[row] = rsqrtf(s / (float)H + 1e-5f); }
    __syncthreads();
    float rstd = rs_s[row];
#pragma unroll
    for (int i = 0; i < 32; ++i) {
        int cc = c0 + 16 * i;
        float hv = (vals[i] - mu) * rstd * hw[cc] + hb[cc];
        hbf[(size_t)(b0 + row) * H + cc] = (__bf16)hv;
        yout[((size_t)(b0 + row) * T + t) * H + cc] = hv;
    }
}

// ---------------------------------------------------------------------------
// in-place row LayerNorm over H=512 (one block = one row, 64 threads)
// ---------------------------------------------------------------------------
__global__ __launch_bounds__(64) void rowln_kernel(
    float* __restrict__ buf, const float* __restrict__ w, const float* __restrict__ b)
{
    const int H = 512;
    size_t row = blockIdx.x;
    int tid = threadIdx.x;
    float* p = buf + row * H;
    __shared__ float sred[64];
    float v[8]; float ps = 0.0f;
#pragma unroll
    for (int e = 0; e < 8; ++e) { int j = tid + 64 * e; v[e] = p[j]; ps += v[e]; }
    sred[tid] = ps; __syncthreads();
    for (int s = 32; s > 0; s >>= 1) { if (tid < s) sred[tid] += sred[tid + s]; __syncthreads(); }
    float mu = sred[0] / (float)H;
    __syncthreads();
    float p2 = 0.0f;
#pragma unroll
    for (int e = 0; e < 8; ++e) { float d = v[e] - mu; p2 += d * d; }
    sred[tid] = p2; __syncthreads();
    for (int s = 32; s > 0; s >>= 1) { if (tid < s) sred[tid] += sred[tid + s]; __syncthreads(); }
    float rstd = rsqrtf(sred[0] / (float)H + 1e-5f);
#pragma unroll
    for (int e = 0; e < 8; ++e) {
        int j = tid + 64 * e;
        p[j] = (v[e] - mu) * rstd * w[j] + b[j];
    }
}

// ---------------------------------------------------------------------------
// prep: f32 -> bf16 convert, f32 copy
// ---------------------------------------------------------------------------
__global__ void cvt_bf16_kernel(const float* __restrict__ src, __bf16* __restrict__ dst, size_t n) {
    size_t i = (size_t)blockIdx.x * blockDim.x + threadIdx.x;
    if (i < n) dst[i] = (__bf16)src[i];
}
__global__ void copyf_kernel(const float* __restrict__ src, float* __restrict__ dst, size_t n) {
    size_t i = (size_t)blockIdx.x * blockDim.x + threadIdx.x;
    if (i < n) dst[i] = src[i];
}

// ---------------------------------------------------------------------------
// host driver
// ---------------------------------------------------------------------------
extern "C" void kernel_launch(void* const* d_in, const int* in_sizes, int n_in,
                              void* d_out, int out_size, void* d_ws, size_t ws_size,
                              hipStream_t stream) {
    (void)in_sizes; (void)n_in; (void)out_size; (void)ws_size;
    const int B = 64, T = 1024, I = 512, H = 512, MM = 2048, L = 2, G = 6656;
    const int K2 = 2 * H;  // 1024

    const float* x        = (const float*)d_in[0];
    const float* proj_w   = (const float*)d_in[1];
    const float* proj_b   = (const float*)d_in[2];
    const float* Wi_w     = (const float*)d_in[3];
    const float* Wi_b     = (const float*)d_in[4];
    const float* Wf_w     = (const float*)d_in[5];
    const float* Wf_b     = (const float*)d_in[6];
    const float* Wo_w     = (const float*)d_in[7];
    const float* Wo_b     = (const float*)d_in[8];
    const float* Wz_w     = (const float*)d_in[9];
    const float* Wz_b     = (const float*)d_in[10];
    const float* Wm_w     = (const float*)d_in[11];
    const float* Wm_b     = (const float*)d_in[12];
    const float* lnc_w    = (const float*)d_in[13];
    const float* lnc_b    = (const float*)d_in[14];
    const float* lnh_w    = (const float*)d_in[15];
    const float* lnh_b    = (const float*)d_in[16];
    const float* out_ln_w = (const float*)d_in[17];
    const float* out_ln_b = (const float*)d_in[18];

    // workspace carve (d_out doubles as the layer-0 input buffer "xproj")
    char* p = (char*)d_ws;
    auto alloc = [&](size_t bytes) -> void* {
        void* r = (void*)p; p += (bytes + 255) & ~(size_t)255; return r;
    };
    float*  ybuf   = (float*)alloc((size_t)B * T * H * 4);
    __bf16* xbf    = (__bf16*)alloc((size_t)B * T * H * 2);   // bf16 layer input
    float*  gates  = (float*)alloc((size_t)B * G * 4);
    float*  cstate = (float*)alloc((size_t)B * MM * 4);
    __bf16* cbf    = (__bf16*)alloc((size_t)B * MM * 2);
    __bf16* hbf    = (__bf16*)alloc((size_t)B * H * 2);
    __bf16* projWb = (__bf16*)alloc((size_t)H * I * 2);
    __bf16* Wcat[2]; __bf16* Wmb[2]; float* cgb[2];
    for (int l = 0; l < L; ++l) {
        Wcat[l] = (__bf16*)alloc((size_t)G * K2 * 2);
        Wmb[l]  = (__bf16*)alloc((size_t)H * MM * 2);
        cgb[l]  = (float*)alloc((size_t)G * 4);
    }
    float* xproj = (float*)d_out;

    auto cdiv = [](size_t a, size_t b) { return (a + b - 1) / b; };
    auto cvt = [&](const float* s, __bf16* d, size_t n) {
        cvt_bf16_kernel<<<dim3((unsigned)cdiv(n, 256)), 256, 0, stream>>>(s, d, n);
    };
    auto cpy = [&](const float* s, float* d, size_t n) {
        copyf_kernel<<<dim3((unsigned)cdiv(n, 256)), 256, 0, stream>>>(s, d, n);
    };

    // ---- prep: bf16 weights + fused gate weight/bias [Wi;Wf;Wo;Wz] ----
    cvt(proj_w, projWb, (size_t)H * I);
    for (int l = 0; l < L; ++l) {
        size_t mo = (size_t)l * MM * K2, ho = (size_t)l * H * K2;
        cvt(Wi_w + mo, Wcat[l],                             (size_t)MM * K2);
        cvt(Wf_w + mo, Wcat[l] + (size_t)MM * K2,           (size_t)MM * K2);
        cvt(Wo_w + ho, Wcat[l] + (size_t)2 * MM * K2,       (size_t)H * K2);
        cvt(Wz_w + mo, Wcat[l] + (size_t)(2 * MM + H) * K2, (size_t)MM * K2);
        cvt(Wm_w + (size_t)l * H * MM, Wmb[l], (size_t)H * MM);
        cpy(Wi_b + (size_t)l * MM, cgb[l],        MM);
        cpy(Wf_b + (size_t)l * MM, cgb[l] + 2048, MM);
        cpy(Wo_b + (size_t)l * H,  cgb[l] + 4096, H);
        cpy(Wz_b + (size_t)l * MM, cgb[l] + 4608, MM);
    }

    // ---- input projection: xproj = x @ proj_w^T + proj_b ----
    {
        int mTiles = (B * T) / 16;          // 4096
        int nGroups = H / (16 * GATE_NT);   // 8
        int blocks = (mTiles * nGroups) / 4;
        wmma_gemm_kernel<<<blocks, 128, 0, stream>>>(
            x, (long long)I, I,
            projWb, (long long)I, proj_b, xproj, (long long)H,
            mTiles, nGroups);
    }

    // ---- layers ----
    for (int l = 0; l < L; ++l) {
        const float* xcur = (l == 0) ? xproj : ybuf;
        float* yo = (l == 0) ? ybuf : (float*)d_out;

        cvt(xcur, xbf, (size_t)B * T * H);                          // layer input -> bf16
        (void)hipMemsetAsync(hbf, 0, (size_t)B * H * 2, stream);    // h = 0
        (void)hipMemsetAsync(cstate, 0, (size_t)B * MM * 4, stream);// c = 0

        for (int t = 0; t < T; ++t) {
            gate_gemm_kernel<<<G / 128, 256, 64 * 1024 * sizeof(__bf16), stream>>>(
                xbf, hbf, Wcat[l], cgb[l], gates, t, T);
            cellln_kernel<<<B, 256, 0, stream>>>(gates, cstate, cbf,
                                                 lnc_w + (size_t)l * MM, lnc_b + (size_t)l * MM);
            cell2_kernel<<<B / 16, 256, 16 * 2048 * sizeof(__bf16), stream>>>(
                cbf, gates, Wmb[l], Wm_b + (size_t)l * H,
                lnh_w + (size_t)l * H, lnh_b + (size_t)l * H,
                hbf, yo, t, T);
        }
        rowln_kernel<<<B * T, 64, 0, stream>>>(yo, out_ln_w + (size_t)l * H, out_ln_b + (size_t)l * H);
    }
}
